// SkillExtractor_64896955843189
// MI455X (gfx1250) — compile-verified, matrix operand order checked by reading, not run on hardware
//
#include <hip/hip_runtime.h>
#include <hip/hip_bf16.h>

// ---------------------------------------------------------------------------
// BiLSTM-CNN-CRF for MI455X (gfx1250): all GEMMs via v_wmma_f32_16x16x32_bf16
// ---------------------------------------------------------------------------

typedef __bf16 bf16_t;
typedef __attribute__((ext_vector_type(16))) __bf16 v16bf;
typedef __attribute__((ext_vector_type(8)))  float  v8f;

#define S_LEN 512
#define BATCH 32
#define EDIM  256
#define CDIM  128
#define HD    256          // per-direction hidden
#define NGATE 1024         // 4*HD
#define TAGS  16
#define MROWS (S_LEN*BATCH)   // 16384

__device__ __forceinline__ v8f wmma_bf16(v16bf a, v16bf b, v8f c) {
  return __builtin_amdgcn_wmma_f32_16x16x32_bf16(false, a, false, b, (short)0, c,
                                                 false, false);
}

// Load a 16x32 bf16 fragment (ISA 7.12.2 layout) from a row-major matrix.
// For the A matrix: rows = M rows.  For B fed as W^T (W row-major [N,K]):
// rows = N columns of B, so per-lane K elements are contiguous in memory.
__device__ __forceinline__ v16bf load_frag16(const bf16_t* base, int ld,
                                             int r0, int k0, int lane) {
  const int r  = lane & 15;
  const int kh = (lane & 16) ? 8 : 0;
  const bf16_t* p = base + (size_t)(r0 + r) * ld + k0 + kh;
  v16bf v;
#pragma unroll
  for (int e = 0; e < 8; ++e) v[e] = p[e];
#pragma unroll
  for (int e = 0; e < 8; ++e) v[8 + e] = p[16 + e];
  return v;
}

__device__ __forceinline__ float sigf(float x) { return 1.f / (1.f + __expf(-x)); }

// --------------------------- prep kernels ----------------------------------

__global__ void k_cvt(const float* __restrict__ src, bf16_t* __restrict__ dst, int n) {
  int i = blockIdx.x * blockDim.x + threadIdx.x;
  if (i < n) dst[i] = (bf16_t)src[i];
}

// conv_w [C,E,3] -> wk[k][C][E] bf16 (contiguous E per (k,c) for B fragments)
__global__ void k_convw(const float* __restrict__ w, bf16_t* __restrict__ wk) {
  int i = blockIdx.x * blockDim.x + threadIdx.x;   // over C*E*3
  if (i >= CDIM * EDIM * 3) return;
  int k  = i % 3;
  int ce = i / 3;                                  // c*E + e
  wk[k * (CDIM * EDIM) + ce] = (bf16_t)w[i];
}

__global__ void k_bias(const float* __restrict__ bi, const float* __restrict__ bh,
                       float* __restrict__ out, int n) {
  int i = blockIdx.x * blockDim.x + threadIdx.x;
  if (i < n) out[i] = bi[i] + bh[i];
}

// embedding gather -> emb_sb[(s*B+b)*E + e], bf16
__global__ void k_embed(const int* __restrict__ x, const float* __restrict__ tab,
                        bf16_t* __restrict__ emb) {
  int idx = blockIdx.x * blockDim.x + threadIdx.x;   // 16384*256 threads
  int e  = idx & (EDIM - 1);
  int sb = idx >> 8;
  int b  = sb & (BATCH - 1);
  int s  = sb >> 5;
  int tok = x[b * S_LEN + s];
  emb[idx] = (bf16_t)tab[(size_t)tok * EDIM + e];
}

// --------------------------- conv as GEMM ----------------------------------
// feat[(s*B+b), c] = relu(conv_b[c] + sum_k W_k[c,:] . emb[(s+k-1)*B+b, :])
// 8192 wave-tiles (1024 m-tiles x 8 n-tiles); one wave per 16x16 tile.
__global__ __launch_bounds__(256) void k_conv(const bf16_t* __restrict__ emb,
                                              const bf16_t* __restrict__ wk,
                                              const float* __restrict__ cb,
                                              bf16_t* __restrict__ feat) {
  int lane = threadIdx.x & 31, wv = threadIdx.x >> 5;
  int tile = blockIdx.x * 8 + wv;
  int mt = tile >> 3, nt = tile & 7;
  int row0 = mt * 16;
  int s = mt >> 1;                         // 16-row tile lies inside one s
  v8f acc = {};
  for (int k = 0; k < 3; ++k) {
    int ss = s + k - 1;
    if (ss < 0 || ss >= S_LEN) continue;   // wave-uniform: EXEC stays full
    const bf16_t* arow = emb + (long)(row0 + (k - 1) * BATCH) * EDIM;
    const bf16_t* wkk  = wk + k * (CDIM * EDIM);
#pragma unroll
    for (int kc = 0; kc < EDIM / 32; ++kc) {
      v16bf a = load_frag16(arow, EDIM, 0, kc * 32, lane);
      v16bf b = load_frag16(wkk, EDIM, nt * 16, kc * 32, lane);
      acc = wmma_bf16(a, b, acc);
    }
  }
  int col = nt * 16 + (lane & 15);
  float bias = cb[col];
  int rbase = (lane & 16) ? 8 : 0;
#pragma unroll
  for (int i = 0; i < 8; ++i) {
    float v = acc[i] + bias;
    v = v > 0.f ? v : 0.f;
    feat[(size_t)(row0 + rbase + i) * CDIM + col] = (bf16_t)v;
  }
}

// ----------------------- LSTM input projection -----------------------------
// xp[dir][m, n] = feat[m,:] . wih_dir[n,:] + (bih+bhh)[n]   (bf16 out)
__global__ __launch_bounds__(256) void k_xproj(const bf16_t* __restrict__ feat,
                                               const bf16_t* __restrict__ wih_f,
                                               const bf16_t* __restrict__ wih_b,
                                               const float* __restrict__ bias_f,
                                               const float* __restrict__ bias_b,
                                               bf16_t* __restrict__ xp) {
  int lane = threadIdx.x & 31, wv = threadIdx.x >> 5;
  int tile = blockIdx.x * 8 + wv;          // 2 * 1024 * 64 tiles
  int dir = tile >> 16;
  int rem = tile & 65535;
  int mt = rem >> 6, nt = rem & 63;
  const bf16_t* w    = dir ? wih_b : wih_f;
  const float*  bias = dir ? bias_b : bias_f;
  v8f acc = {};
#pragma unroll
  for (int kc = 0; kc < CDIM / 32; ++kc) {
    v16bf a = load_frag16(feat, CDIM, mt * 16, kc * 32, lane);
    v16bf b = load_frag16(w, CDIM, nt * 16, kc * 32, lane);
    acc = wmma_bf16(a, b, acc);
  }
  int col = nt * 16 + (lane & 15);
  float bv = bias[col];
  int rbase = (lane & 16) ? 8 : 0;
  bf16_t* out = xp + (size_t)dir * MROWS * NGATE;
#pragma unroll
  for (int i = 0; i < 8; ++i)
    out[(size_t)(mt * 16 + rbase + i) * NGATE + col] = (bf16_t)(acc[i] + bv);
}

// --------------------------- LSTM recurrence -------------------------------
// One persistent workgroup per direction.  16 waves; wave w owns hidden
// columns [w*16, w*16+16).  h lives in LDS (bf16), cell state in registers.
__global__ __launch_bounds__(512) void k_lstm(const bf16_t* __restrict__ xp_all,
                                              const bf16_t* __restrict__ whh_f,
                                              const bf16_t* __restrict__ whh_b,
                                              bf16_t* __restrict__ hcat) {
  __shared__ bf16_t hsh[BATCH * HD];       // 16 KB
  int dir = blockIdx.x;
  const bf16_t* xp  = xp_all + (size_t)dir * MROWS * NGATE;
  const bf16_t* whh = dir ? whh_b : whh_f;
  int lane = threadIdx.x & 31, wv = threadIdx.x >> 5;   // wv in 0..15
  for (int i = threadIdx.x; i < BATCH * HD; i += 512) hsh[i] = (bf16_t)0.f;
  __syncthreads();

  int jc = wv * 16;
  int dcol = lane & 15;
  int rbase = (lane & 16) ? 8 : 0;
  v8f cst[2] = {v8f{}, v8f{}};             // cell-state fragments (rows 0-15/16-31)

  for (int t = 0; t < S_LEN; ++t) {
    int s = dir ? (S_LEN - 1 - t) : t;
    size_t row0 = (size_t)s * BATCH;
    v8f acc[2][4];
    // init accumulators from precomputed x-projection (bias folded in)
#pragma unroll
    for (int m2 = 0; m2 < 2; ++m2)
#pragma unroll
      for (int g = 0; g < 4; ++g) {
        int gcol = g * HD + jc + dcol;
#pragma unroll
        for (int i = 0; i < 8; ++i)
          acc[m2][g][i] = (float)xp[(row0 + m2 * 16 + rbase + i) * NGATE + gcol];
      }
    // gates += h . whh^T  (K = 256, 8 chunks; 64 WMMAs per wave per step)
#pragma unroll
    for (int g = 0; g < 4; ++g) {
#pragma unroll
      for (int kc = 0; kc < HD / 32; ++kc) {
        v16bf bfrag = load_frag16(whh, HD, g * HD + jc, kc * 32, lane);
        v16bf a0 = load_frag16(hsh, HD, 0, kc * 32, lane);
        v16bf a1 = load_frag16(hsh, HD, 16, kc * 32, lane);
        acc[0][g] = wmma_bf16(a0, bfrag, acc[0][g]);
        acc[1][g] = wmma_bf16(a1, bfrag, acc[1][g]);
      }
    }
    __syncthreads();                       // all hsh reads complete
#pragma unroll
    for (int m2 = 0; m2 < 2; ++m2) {
#pragma unroll
      for (int i = 0; i < 8; ++i) {
        float ig = sigf(acc[m2][0][i]);
        float fg = sigf(acc[m2][1][i]);
        float gg = tanhf(acc[m2][2][i]);
        float og = sigf(acc[m2][3][i]);
        float c = fg * cst[m2][i] + ig * gg;
        cst[m2][i] = c;
        float h = og * tanhf(c);
        int row = m2 * 16 + rbase + i;
        hsh[row * HD + jc + dcol] = (bf16_t)h;
        hcat[(row0 + row) * (2 * HD) + dir * HD + jc + dcol] = (bf16_t)h;
      }
    }
    __syncthreads();                       // new h visible before next step
  }
}

// ------------------------------ FC (emissions) -----------------------------
__global__ __launch_bounds__(256) void k_fc(const bf16_t* __restrict__ hcat,
                                            const bf16_t* __restrict__ fcw,
                                            const float* __restrict__ fcb,
                                            float* __restrict__ emis) {
  int lane = threadIdx.x & 31, wv = threadIdx.x >> 5;
  int mt = blockIdx.x * 8 + wv;            // 1024 m-tiles, N = 16 (one tile)
  v8f acc = {};
#pragma unroll
  for (int kc = 0; kc < (2 * HD) / 32; ++kc) {
    v16bf a = load_frag16(hcat, 2 * HD, mt * 16, kc * 32, lane);
    v16bf b = load_frag16(fcw, 2 * HD, 0, kc * 32, lane);
    acc = wmma_bf16(a, b, acc);
  }
  int col = lane & 15;
  float bv = fcb[col];
  int rbase = (lane & 16) ? 8 : 0;
#pragma unroll
  for (int i = 0; i < 8; ++i)
    emis[(size_t)(mt * 16 + rbase + i) * TAGS + col] = acc[i] + bv;
}

// ------------------------------ CRF NLL ------------------------------------
// emis layout [S, B, T]; tags/mask [B, S].  One block per batch element.
__global__ void k_crf(const float* __restrict__ emis, const int* __restrict__ tags,
                      const int* __restrict__ mask, const float* __restrict__ st,
                      const float* __restrict__ et, const float* __restrict__ tr,
                      float* __restrict__ res) {
  int b = blockIdx.x;
  int t = threadIdx.x;
  __shared__ float trs[TAGS][TAGS];
  __shared__ float alpha[TAGS], nxt[TAGS];
  __shared__ float sscore;
  if (t < TAGS) {
    for (int p = 0; p < TAGS; ++p) trs[p][t] = tr[p * TAGS + t];
    alpha[t] = st[t] + emis[(size_t)b * TAGS + t];
  }
  __syncthreads();
  if (t == 0) {                            // gold-path score (serial, tiny)
    int prev = tags[b * S_LEN];
    float score = st[prev] + emis[(size_t)b * TAGS + prev];
    int send = 0;
    for (int s2 = 0; s2 < S_LEN; ++s2) send += mask[b * S_LEN + s2];
    send -= 1;
    for (int s2 = 1; s2 < S_LEN; ++s2) {
      int cur = tags[b * S_LEN + s2];
      float mi = (float)mask[b * S_LEN + s2];
      score += (tr[prev * TAGS + cur] +
                emis[((size_t)s2 * BATCH + b) * TAGS + cur]) * mi;
      prev = cur;
    }
    score += et[tags[b * S_LEN + send]];
    sscore = score;
  }
  for (int s2 = 1; s2 < S_LEN; ++s2) {     // forward algorithm
    float mi = (float)mask[b * S_LEN + s2];
    if (t < TAGS) {
      float em = emis[((size_t)s2 * BATCH + b) * TAGS + t];
      float mx = -1e30f;
      for (int p = 0; p < TAGS; ++p) {
        float v = alpha[p] + trs[p][t];
        mx = mx > v ? mx : v;
      }
      float sum = 0.f;
      for (int p = 0; p < TAGS; ++p) sum += __expf(alpha[p] + trs[p][t] - mx);
      float nv = mx + __logf(sum) + em;
      nxt[t] = (mi > 0.f) ? nv : alpha[t];
    }
    __syncthreads();
    if (t < TAGS) alpha[t] = nxt[t];
    __syncthreads();
  }
  if (t == 0) {
    float mx = -1e30f;
    for (int p = 0; p < TAGS; ++p) {
      float v = alpha[p] + et[p];
      mx = mx > v ? mx : v;
    }
    float sum = 0.f;
    for (int p = 0; p < TAGS; ++p) sum += __expf(alpha[p] + et[p] - mx);
    res[b] = (mx + __logf(sum)) - sscore;  // logz - score
  }
}

__global__ void k_mean(const float* __restrict__ res, float* __restrict__ out) {
  if (threadIdx.x == 0) {
    float s = 0.f;
    for (int i = 0; i < BATCH; ++i) s += res[i];
    out[0] = s / (float)BATCH;             // -mean(score - logz)
  }
}

// ---------------------------------------------------------------------------

extern "C" void kernel_launch(void* const* d_in, const int* in_sizes, int n_in,
                              void* d_out, int out_size, void* d_ws, size_t ws_size,
                              hipStream_t stream) {
  (void)in_sizes; (void)n_in; (void)out_size; (void)ws_size;
  const int*   x        = (const int*)d_in[0];
  const int*   maskp    = (const int*)d_in[1];
  const int*   tagsp    = (const int*)d_in[2];
  const float* emb_tab  = (const float*)d_in[3];
  const float* conv_w   = (const float*)d_in[4];
  const float* conv_b   = (const float*)d_in[5];
  const float* wih_f    = (const float*)d_in[6];
  const float* whh_f    = (const float*)d_in[7];
  const float* bih_f    = (const float*)d_in[8];
  const float* bhh_f    = (const float*)d_in[9];
  const float* wih_b    = (const float*)d_in[10];
  const float* whh_b    = (const float*)d_in[11];
  const float* bih_b    = (const float*)d_in[12];
  const float* bhh_b    = (const float*)d_in[13];
  const float* fc_w     = (const float*)d_in[14];
  const float* fc_b     = (const float*)d_in[15];
  const float* start_tr = (const float*)d_in[16];
  const float* end_tr   = (const float*)d_in[17];
  const float* trans    = (const float*)d_in[18];
  float* out = (float*)d_out;

  char* w = (char*)d_ws;
  size_t off = 0;
  auto alloc = [&](size_t bytes) -> void* {
    void* p = w + off;
    off += (bytes + 255) & ~(size_t)255;
    return p;
  };
  bf16_t* emb_sb  = (bf16_t*)alloc((size_t)MROWS * EDIM * 2);        // 8 MB
  bf16_t* feat    = (bf16_t*)alloc((size_t)MROWS * CDIM * 2);        // 4 MB
  bf16_t* wk      = (bf16_t*)alloc((size_t)3 * CDIM * EDIM * 2);
  bf16_t* wihf_bf = (bf16_t*)alloc((size_t)NGATE * CDIM * 2);
  bf16_t* wihb_bf = (bf16_t*)alloc((size_t)NGATE * CDIM * 2);
  bf16_t* whhf_bf = (bf16_t*)alloc((size_t)NGATE * HD * 2);
  bf16_t* whhb_bf = (bf16_t*)alloc((size_t)NGATE * HD * 2);
  bf16_t* fcw_bf  = (bf16_t*)alloc((size_t)TAGS * 2 * HD * 2);
  float*  bias_f  = (float*)alloc((size_t)NGATE * 4);
  float*  bias_b  = (float*)alloc((size_t)NGATE * 4);
  bf16_t* xproj   = (bf16_t*)alloc((size_t)2 * MROWS * NGATE * 2);   // 64 MB
  bf16_t* hcat    = (bf16_t*)alloc((size_t)MROWS * 2 * HD * 2);      // 16 MB
  float*  emis    = (float*)alloc((size_t)MROWS * TAGS * 4);         // 1 MB
  float*  resb    = (float*)alloc((size_t)BATCH * 4);

  // weight conversion / packing
  k_cvt<<<(NGATE * CDIM + 255) / 256, 256, 0, stream>>>(wih_f, wihf_bf, NGATE * CDIM);
  k_cvt<<<(NGATE * CDIM + 255) / 256, 256, 0, stream>>>(wih_b, wihb_bf, NGATE * CDIM);
  k_cvt<<<(NGATE * HD + 255) / 256, 256, 0, stream>>>(whh_f, whhf_bf, NGATE * HD);
  k_cvt<<<(NGATE * HD + 255) / 256, 256, 0, stream>>>(whh_b, whhb_bf, NGATE * HD);
  k_cvt<<<(TAGS * 2 * HD + 255) / 256, 256, 0, stream>>>(fc_w, fcw_bf, TAGS * 2 * HD);
  k_convw<<<(CDIM * EDIM * 3 + 255) / 256, 256, 0, stream>>>(conv_w, wk);
  k_bias<<<(NGATE + 255) / 256, 256, 0, stream>>>(bih_f, bhh_f, bias_f, NGATE);
  k_bias<<<(NGATE + 255) / 256, 256, 0, stream>>>(bih_b, bhh_b, bias_b, NGATE);

  // pipeline
  k_embed<<<(MROWS * EDIM) / 256, 256, 0, stream>>>(x, emb_tab, emb_sb);
  k_conv<<<(MROWS / 16) * (CDIM / 16) / 8, 256, 0, stream>>>(emb_sb, wk, conv_b, feat);
  k_xproj<<<2 * (MROWS / 16) * (NGATE / 16) / 8, 256, 0, stream>>>(
      feat, wihf_bf, wihb_bf, bias_f, bias_b, xproj);
  k_lstm<<<2, 512, 0, stream>>>(xproj, whhf_bf, whhb_bf, hcat);
  k_fc<<<(MROWS / 16) / 8, 256, 0, stream>>>(hcat, fcw_bf, fc_b, emis);
  k_crf<<<BATCH, 32, 0, stream>>>(emis, tagsp, maskp, start_tr, end_tr, trans, resb);
  k_mean<<<1, 32, 0, stream>>>(resb, out);
}
